// Encoder_48765058679430
// MI455X (gfx1250) — compile-verified, matrix operand order checked by reading, not run on hardware
//
#include <hip/hip_runtime.h>

typedef __bf16 bf16;
typedef __attribute__((ext_vector_type(16))) __bf16 v16bf;
typedef __attribute__((ext_vector_type(8)))  float  v8f;
typedef __attribute__((ext_vector_type(4)))  int    v4i;

#define LRELU_K 0.1f
#define EPS_K   1e-5f

#if defined(__HIP_DEVICE_COMPILE__) && \
    __has_builtin(__builtin_amdgcn_global_load_async_to_lds_b128) && \
    __has_builtin(__builtin_amdgcn_s_wait_asynccnt)
#define USE_ASYNC_LDS 1
#else
#define USE_ASYNC_LDS 0
#endif

// ---------------------------------------------------------------------------
// Cooperative copy of nchunks*16B from contiguous global into contiguous LDS,
// via gfx1250 async global->LDS DMA when available (tracked on ASYNCcnt).
// Builtin signature (from toolchain): (v4i addrspace(1)*, v4i addrspace(3)*,
// imm offset, imm cpol).
// ---------------------------------------------------------------------------
__device__ __forceinline__ void stage_chunks(bf16* lds_dst, const bf16* gsrc, int nchunks) {
  for (int c = threadIdx.x; c < nchunks; c += 256) {
#if USE_ASYNC_LDS
    __builtin_amdgcn_global_load_async_to_lds_b128(
        (__attribute__((address_space(1))) v4i*)(gsrc + (size_t)c * 8),
        (__attribute__((address_space(3))) v4i*)(lds_dst + (size_t)c * 8),
        0, 0);
#else
    *(uint4*)(lds_dst + (size_t)c * 8) = *(const uint4*)(gsrc + (size_t)c * 8);
#endif
  }
}
__device__ __forceinline__ void stage_wait() {
#if USE_ASYNC_LDS
  __builtin_amdgcn_s_wait_asynccnt(0);
#endif
  __syncthreads();
}

// ---------------------------------------------------------------------------
// WMMA fragment load for a row-major 16-bit matrix, A-operand layout
// (also valid for the B operand of C = A * B^T, since B rows are the N index).
// 16-bit 16x32 A layout: lanes 0-15 -> M=lane, K in {0..7, 16..23};
//                        lanes 16-31 -> M=lane-16, K in {8..15, 24..31}.
// Works on global or LDS (generic) pointers; emits b128/ds_load_b128 pairs.
// ---------------------------------------------------------------------------
__device__ __forceinline__ v16bf ldfrag(const bf16* base, int ld, int row0, int k0) {
  const int lane = threadIdx.x & 31;
  const bf16* p = base + (size_t)(row0 + (lane & 15)) * ld + k0 + ((lane >> 4) ? 8 : 0);
  v16bf f;
#pragma unroll
  for (int i = 0; i < 8; ++i) { f[i] = p[i]; f[8 + i] = p[16 + i]; }
  return f;
}

// One 16x16 fp32 C tile of  C = A(MxK) * B(NxK)^T, both row-major bf16.
__device__ __forceinline__ v8f wmma_abT(const bf16* A, int lda, const bf16* B, int ldb,
                                        int m0, int n0, int K) {
  v8f acc = {};
  for (int k0 = 0; k0 < K; k0 += 32) {
    v16bf a = ldfrag(A, lda, m0, k0);
    v16bf b = ldfrag(B, ldb, n0, k0);
    acc = __builtin_amdgcn_wmma_f32_16x16x32_bf16(false, a, false, b,
                                                  (short)0, acc, false, false);
  }
  return acc;
}

// ---------------------------------------------------------------------------
// Big input-projection GEMM: gx = Xe(M x 512) * Wt(1536 x 512)^T + bx[n]
// grid = (N/16, M/128), block = 256. Shared B tile staged to LDS once/block.
// ---------------------------------------------------------------------------
__global__ void gx_gemm(const bf16* __restrict__ A, const bf16* __restrict__ Wt,
                        const float* __restrict__ bx, float* __restrict__ C,
                        int N, int K) {
  __shared__ __align__(16) bf16 bsm[16 * 512];
  const int wave = threadIdx.x >> 5, lane = threadIdx.x & 31;
  const int n0 = blockIdx.x * 16;
  const int m0 = (blockIdx.y * 8 + wave) * 16;
  stage_chunks(bsm, Wt + (size_t)n0 * K, 2 * K);   // 16 rows * K bf16 = 2K chunks
  stage_wait();
  v8f acc = wmma_abT(A, K, bsm, K, m0, 0, K);
  const int col = n0 + (lane & 15);
  const int rbase = m0 + ((lane >> 4) ? 8 : 0);
  const float bb = bx[col];
#pragma unroll
  for (int j = 0; j < 8; ++j)
    C[(size_t)(rbase + j) * N + col] = acc[j] + bb;
}

// ---------------------------------------------------------------------------
// One GRU time step, fully fused. The 3 gate weight tiles (48 rows x 512)
// are staged into LDS once per block; each A fragment is loaded once and
// used by 3 WMMAs (z, r, hhat). grid = 32 blocks, block = 256 (8 waves).
// ---------------------------------------------------------------------------
__global__ void gru_step(const bf16* __restrict__ Aprev, int lda,
                         const bf16* __restrict__ Wht, const float* __restrict__ bh,
                         const float* __restrict__ gx,     // row base t*B, ld=1536
                         const float* __restrict__ maskrow,
                         float* __restrict__ h,            // (B,512) fp32 state
                         bf16* __restrict__ outb, int ldob,
                         float* __restrict__ outf) {       // nullable, ld=512
  __shared__ __align__(16) bf16 wsm[48 * 512];             // 48 KB
  const int wave = threadIdx.x >> 5, lane = threadIdx.x & 31;
  const int e0 = blockIdx.x * 16;
  const int m0 = wave * 16;
#pragma unroll
  for (int g = 0; g < 3; ++g)
    stage_chunks(wsm + g * 16 * 512, Wht + (size_t)(g * 512 + e0) * 512, 1024);
  stage_wait();

  v8f az = {}, ar = {}, ah = {};
  for (int k0 = 0; k0 < 512; k0 += 32) {
    v16bf a  = ldfrag(Aprev, lda, m0, k0);
    v16bf b0 = ldfrag(wsm, 512, 0,  k0);
    v16bf b1 = ldfrag(wsm, 512, 16, k0);
    v16bf b2 = ldfrag(wsm, 512, 32, k0);
    az = __builtin_amdgcn_wmma_f32_16x16x32_bf16(false, a, false, b0, (short)0, az, false, false);
    ar = __builtin_amdgcn_wmma_f32_16x16x32_bf16(false, a, false, b1, (short)0, ar, false, false);
    ah = __builtin_amdgcn_wmma_f32_16x16x32_bf16(false, a, false, b2, (short)0, ah, false, false);
  }

  const int e = e0 + (lane & 15);
  const int rbase = m0 + ((lane >> 4) ? 8 : 0);
  const float bhz = bh[e], bhr = bh[512 + e], bhh = bh[1024 + e];
#pragma unroll
  for (int j = 0; j < 8; ++j) {
    const int b = rbase + j;
    const float* gxr = gx + (size_t)b * 1536;
    float z = 1.f / (1.f + __expf(-(gxr[e] + az[j] + bhz)));
    float r = 1.f / (1.f + __expf(-(gxr[512 + e] + ar[j] + bhr)));
    float hh = tanhf(gxr[1024 + e] + r * (ah[j] + bhh));
    float hp = h[(size_t)b * 512 + e];
    float m  = maskrow[b];
    float hn = (1.f - z) * hp + z * hh;
    hn = m * hn + (1.f - m) * hp;
    h[(size_t)b * 512 + e] = hn;
    outb[(size_t)b * ldob + e] = (bf16)hn;
    if (outf) outf[(size_t)b * 512 + e] = hn;
  }
}

// ---------------------------------------------------------------------------
// Batched C = A * B^T with optional per-row bias, leaky-relu, f32 or bf16 out.
// All call sites have ldb == K, so the block's 16xK B tile is contiguous and
// staged into LDS once. grid = (N/16, Mtiles/8, batch), block = 256.
// ---------------------------------------------------------------------------
__global__ void bgemm_abT(const bf16* __restrict__ A, long long sA, int lda,
                          const bf16* __restrict__ Bm, long long sB, int ldb,
                          float* __restrict__ Cf, bf16* __restrict__ Cb,
                          long long sC, int ldc,
                          const float* __restrict__ bias, int leaky, int K) {
  __shared__ __align__(16) bf16 bsm[16 * 512];
  const int wave = threadIdx.x >> 5, lane = threadIdx.x & 31;
  const int bz = blockIdx.z;
  const bf16* Ab = A + (size_t)bz * sA;
  const bf16* Bb = Bm + (size_t)bz * sB;
  const int n0 = blockIdx.x * 16;
  const int m0 = (blockIdx.y * 8 + wave) * 16;
  stage_chunks(bsm, Bb + (size_t)n0 * ldb, 2 * K);
  stage_wait();
  v8f acc = wmma_abT(Ab, lda, bsm, K, m0, 0, K);
  const int col = n0 + (lane & 15);
  const int rbase = m0 + ((lane >> 4) ? 8 : 0);
#pragma unroll
  for (int j = 0; j < 8; ++j) {
    const int rr = rbase + j;
    float v = acc[j];
    if (bias) v += bias[rr];
    if (leaky) v = (v >= 0.f) ? v : LRELU_K * v;
    const size_t idx = (size_t)bz * sC + (size_t)rr * ldc + col;
    if (Cf) Cf[idx] = v;
    else    Cb[idx] = (bf16)v;
  }
}

// ---------------------------------------------------------------------------
// Masked softmax over rows of f (B,Lp,Lp); one wave per row, bf16 output.
// ---------------------------------------------------------------------------
__global__ void softmax_rows(const float* __restrict__ f, const float* __restrict__ xs_mask,
                             bf16* __restrict__ fb, int Bn, int Lp, int L) {
  const int wave = threadIdx.x >> 5, lane = threadIdx.x & 31;
  const int row = blockIdx.x * 8 + wave;          // row index in [0, Bn*Lp)
  const int b = row / Lp;
  const float* s = f + (size_t)row * Lp;
  float vals[4], msk[4];
  float mx = -1e30f;
#pragma unroll
  for (int i = 0; i < 4; ++i) {
    const int m = lane + 32 * i;
    vals[i] = s[m];
    msk[i] = (m < L) ? xs_mask[(size_t)m * Bn + b] : 0.f;
    mx = fmaxf(mx, vals[i]);
  }
  for (int o = 16; o > 0; o >>= 1) mx = fmaxf(mx, __shfl_xor(mx, o, 32));
  float sum = 0.f;
#pragma unroll
  for (int i = 0; i < 4; ++i) { vals[i] = __expf(vals[i] - mx); sum += vals[i] * msk[i]; }
  for (int o = 16; o > 0; o >>= 1) sum += __shfl_xor(sum, o, 32);
  const float inv = 1.f / sum;
#pragma unroll
  for (int i = 0; i < 4; ++i)
    fb[(size_t)row * Lp + lane + 32 * i] = (bf16)(vals[i] * inv * msk[i]);
}

// ---------------------------------------------------------------------------
// BatchNorm1d stats over (B, l<L) per channel; x is (B, C, Lp) fp32.
// ---------------------------------------------------------------------------
__global__ void bn_stats(const float* __restrict__ x, int C, int Lp, int L, int Bn,
                         float* __restrict__ mean, float* __restrict__ invstd) {
  const int c = blockIdx.x;
  float s = 0.f, s2 = 0.f;
  for (int i = threadIdx.x; i < Bn * L; i += blockDim.x) {
    const int b = i / L, l = i % L;
    const float v = x[((size_t)b * C + c) * Lp + l];
    s += v; s2 += v * v;
  }
  __shared__ float sh0[256], sh1[256];
  sh0[threadIdx.x] = s; sh1[threadIdx.x] = s2;
  __syncthreads();
  for (int o = 128; o > 0; o >>= 1) {
    if ((int)threadIdx.x < o) { sh0[threadIdx.x] += sh0[threadIdx.x + o];
                                sh1[threadIdx.x] += sh1[threadIdx.x + o]; }
    __syncthreads();
  }
  if (threadIdx.x == 0) {
    const float n = (float)(Bn * L);
    const float mu = sh0[0] / n;
    const float var = sh1[0] / n - mu * mu;
    mean[c] = mu; invstd[c] = rsqrtf(var + EPS_K);
  }
}

__global__ void bn_apply_bf(const float* __restrict__ x, int C, int Lp, int L,
                            const float* __restrict__ mean, const float* __restrict__ invstd,
                            const float* __restrict__ gamma, const float* __restrict__ beta,
                            bf16* __restrict__ out, size_t n) {
  const size_t i = (size_t)blockIdx.x * blockDim.x + threadIdx.x;
  if (i >= n) return;
  const int l = (int)(i % Lp);
  const int c = (int)((i / Lp) % C);
  float v = 0.f;
  if (l < L) v = gamma[c] * (x[i] - mean[c]) * invstd[c] + beta[c];
  out[i] = (bf16)v;
}

// final: out[l,b,d] = bn(ylast_pre)[b,d,l] + enc_f32[l,b,d]
__global__ void final_out(const float* __restrict__ ylp, const float* __restrict__ mean,
                          const float* __restrict__ invstd, const float* __restrict__ gamma,
                          const float* __restrict__ beta, const float* __restrict__ encf,
                          float* __restrict__ out, int Bn, int D, int Lp, size_t n) {
  const size_t i = (size_t)blockIdx.x * blockDim.x + threadIdx.x;
  if (i >= n) return;
  const int d = (int)(i % D);
  const int b = (int)((i / D) % Bn);
  const int l = (int)(i / ((size_t)D * Bn));
  const float v = gamma[d] * (ylp[((size_t)b * D + d) * Lp + l] - mean[d]) * invstd[d] + beta[d];
  out[i] = v + encf[i];
}

// ---------------------------------------------------------------------------
// Prep kernels
// ---------------------------------------------------------------------------
// (3,512,512) fp32 -> (1536,512) bf16 transposed: out[n][d] = W[n/512][d][n%512]
__global__ void pack_w3(const float* __restrict__ W, bf16* __restrict__ out, size_t n) {
  const size_t i = (size_t)blockIdx.x * blockDim.x + threadIdx.x;
  if (i >= n) return;
  const int d = (int)(i % 512);
  const int nn = (int)(i / 512);
  const int k = nn >> 9, e = nn & 511;
  out[i] = (bf16)W[((size_t)k * 512 + d) * 512 + e];
}

__global__ void cvt_bf(const float* __restrict__ in, bf16* __restrict__ out, size_t n) {
  const size_t i = (size_t)blockIdx.x * blockDim.x + threadIdx.x;
  if (i < n) out[i] = (bf16)in[i];
}

__global__ void zero_bf(bf16* __restrict__ p, size_t n) {
  const size_t i = (size_t)blockIdx.x * blockDim.x + threadIdx.x;
  if (i < n) p[i] = (bf16)0.f;
}
__global__ void zero_f32(float* __restrict__ p, size_t n) {
  const size_t i = (size_t)blockIdx.x * blockDim.x + threadIdx.x;
  if (i < n) p[i] = 0.f;
}

__global__ void gather_emb(const int* __restrict__ xs, const float* __restrict__ emb,
                           bf16* __restrict__ out, size_t n) {
  const size_t i = (size_t)blockIdx.x * blockDim.x + threadIdx.x;
  if (i >= n) return;
  const int d = (int)(i % 512);
  const size_t rb = i / 512;                 // t*B + b
  const int idx = xs[rb];
  out[i] = (bf16)emb[(size_t)idx * 512 + d];
}

// ---------------------------------------------------------------------------
extern "C" void kernel_launch(void* const* d_in, const int* in_sizes, int n_in,
                              void* d_out, int out_size, void* d_ws, size_t ws_size,
                              hipStream_t stream) {
  const int L = 100, B = 128, D = 512, Lp = 128, N3 = 1536, C2 = 256;

  const int*   xs      = (const int*)  d_in[0];
  const float* xs_mask = (const float*)d_in[1];
  const float* emb     = (const float*)d_in[2];
  const float* fw_Wx   = (const float*)d_in[3];
  const float* fw_Wh   = (const float*)d_in[4];
  const float* fw_bx   = (const float*)d_in[5];
  const float* fw_bh   = (const float*)d_in[6];
  const float* bw_Wx   = (const float*)d_in[7];
  const float* bw_Wh   = (const float*)d_in[8];
  const float* bw_bx   = (const float*)d_in[9];
  const float* bw_bh   = (const float*)d_in[10];
  const float* Wg      = (const float*)d_in[11];
  const float* bg      = (const float*)d_in[12];
  const float* g_gamma = (const float*)d_in[13];
  const float* g_beta  = (const float*)d_in[14];
  const float* Wlast   = (const float*)d_in[15];
  const float* blast   = (const float*)d_in[16];
  const float* l_gamma = (const float*)d_in[17];
  const float* l_beta  = (const float*)d_in[18];
  float* out = (float*)d_out;

  char* p = (char*)d_ws;
  auto carve = [&](size_t bytes) -> char* {
    char* r = p; p += (bytes + 255) & ~(size_t)255; return r;
  };
  bf16* wxt_fw   = (bf16*)carve((size_t)N3 * D * 2);
  bf16* wht_fw   = (bf16*)carve((size_t)N3 * D * 2);
  bf16* wxt_bw   = (bf16*)carve((size_t)N3 * D * 2);
  bf16* wht_bw   = (bf16*)carve((size_t)N3 * D * 2);
  bf16* wg_bf    = (bf16*)carve((size_t)C2 * D * 2);
  bf16* wlast_bf = (bf16*)carve((size_t)D * C2 * 2);
  bf16* xs_e_bf  = (bf16*)carve((size_t)L * B * D * 2);
  float* gxbuf   = (float*)carve((size_t)L * B * N3 * 4);     // reused fw then bw
  bf16* hs_bf    = (bf16*)carve((size_t)L * B * D * 2);       // fw hidden states
  float* h_fw    = (float*)carve((size_t)B * D * 4);
  float* h_bw    = (float*)carve((size_t)B * D * 4);
  bf16* h0_bf    = (bf16*)carve((size_t)B * D * 2);
  float* enc_f32 = (float*)carve((size_t)L * B * D * 4);      // (L,B,D)
  bf16* enc_bf   = (bf16*)carve((size_t)B * Lp * D * 2);      // (B,Lp,D), pads zeroed
  float* f_f32   = (float*)carve((size_t)B * Lp * Lp * 4);
  bf16* f_bf     = (bf16*)carve((size_t)B * Lp * Lp * 2);
  float* g_pre   = (float*)carve((size_t)B * C2 * Lp * 4);    // (B,256,Lp)
  bf16* g_bf     = (bf16*)carve((size_t)B * C2 * Lp * 2);
  bf16* y_bf     = (bf16*)carve((size_t)B * Lp * C2 * 2);     // (B,Lp,256)
  float* ylast   = (float*)carve((size_t)B * D * Lp * 4);     // (B,512,Lp)
  float* mean_g  = (float*)carve(C2 * 4);
  float* inv_g   = (float*)carve(C2 * 4);
  float* mean_l  = (float*)carve(D * 4);
  float* inv_l   = (float*)carve(D * 4);

  auto gsz = [](size_t n) { return (unsigned)((n + 255) / 256); };

  // ---- weight prep ----
  { size_t n = (size_t)N3 * D;
    pack_w3<<<gsz(n), 256, 0, stream>>>(fw_Wx, wxt_fw, n);
    pack_w3<<<gsz(n), 256, 0, stream>>>(fw_Wh, wht_fw, n);
    pack_w3<<<gsz(n), 256, 0, stream>>>(bw_Wx, wxt_bw, n);
    pack_w3<<<gsz(n), 256, 0, stream>>>(bw_Wh, wht_bw, n); }
  { size_t n = (size_t)C2 * D;
    cvt_bf<<<gsz(n), 256, 0, stream>>>(Wg, wg_bf, n);
    cvt_bf<<<gsz(n), 256, 0, stream>>>(Wlast, wlast_bf, n); }

  // ---- embedding gather + state init ----
  { size_t n = (size_t)L * B * D;
    gather_emb<<<gsz(n), 256, 0, stream>>>(xs, emb, xs_e_bf, n); }
  { size_t n = (size_t)B * D;
    zero_f32<<<gsz(n), 256, 0, stream>>>(h_fw, n);
    zero_f32<<<gsz(n), 256, 0, stream>>>(h_bw, n);
    zero_bf <<<gsz(n), 256, 0, stream>>>(h0_bf, n); }
  { size_t n = (size_t)B * Lp * D;
    zero_bf<<<gsz(n), 256, 0, stream>>>(enc_bf, n); }

  // ---- forward GRU: gx = Xe * Wxt^T, then 100 fused steps ----
  gx_gemm<<<dim3(N3 / 16, (L * B) / 128), 256, 0, stream>>>(xs_e_bf, wxt_fw, fw_bx, gxbuf, N3, D);
  for (int t = 0; t < L; ++t) {
    const bf16* Aprev = (t == 0) ? h0_bf : hs_bf + (size_t)(t - 1) * B * D;
    gru_step<<<D / 16, 256, 0, stream>>>(Aprev, D, wht_fw, fw_bh,
                                         gxbuf + (size_t)t * B * N3,
                                         xs_mask + (size_t)t * B,
                                         h_fw, hs_bf + (size_t)t * B * D, D, nullptr);
  }

  // ---- backward GRU over reversed right ----
  gx_gemm<<<dim3(N3 / 16, (L * B) / 128), 256, 0, stream>>>(hs_bf, wxt_bw, bw_bx, gxbuf, N3, D);
  for (int t = 0; t < L; ++t) {
    const int l = L - 1 - t;
    const bf16* Aprev = (t == 0) ? h0_bf : enc_bf + (size_t)(l + 1) * D;
    const int lda = (t == 0) ? D : Lp * D;
    gru_step<<<D / 16, 256, 0, stream>>>(Aprev, lda, wht_bw, bw_bh,
                                         gxbuf + (size_t)l * B * N3,
                                         xs_mask + (size_t)l * B,
                                         h_bw, enc_bf + (size_t)l * D, Lp * D,
                                         enc_f32 + (size_t)l * B * D);
  }

  // ---- non-local block ----
  // f = enc_b * enc_b^T  (per batch, 128x128, K=512)
  bgemm_abT<<<dim3(Lp / 16, 1, B), 256, 0, stream>>>(
      enc_bf, (long long)Lp * D, D, enc_bf, (long long)Lp * D, D,
      f_f32, nullptr, (long long)Lp * Lp, Lp, nullptr, 0, D);
  softmax_rows<<<(B * Lp) / 8, 256, 0, stream>>>(f_f32, xs_mask, f_bf, B, Lp, L);

  // g_pre[b,c,l] = leaky(Wg * enc_b^T + bg)   (256 x 128, K=512)
  bgemm_abT<<<dim3(Lp / 16, C2 / 128, B), 256, 0, stream>>>(
      wg_bf, 0, D, enc_bf, (long long)Lp * D, D,
      g_pre, nullptr, (long long)C2 * Lp, Lp, bg, 1, D);
  bn_stats<<<C2, 256, 0, stream>>>(g_pre, C2, Lp, L, B, mean_g, inv_g);
  { size_t n = (size_t)B * C2 * Lp;
    bn_apply_bf<<<gsz(n), 256, 0, stream>>>(g_pre, C2, Lp, L, mean_g, inv_g,
                                            g_gamma, g_beta, g_bf, n); }

  // y[b,l,c] = f * g^T   (128 x 256, K=128) -> bf16
  bgemm_abT<<<dim3(C2 / 16, 1, B), 256, 0, stream>>>(
      f_bf, (long long)Lp * Lp, Lp, g_bf, (long long)C2 * Lp, Lp,
      nullptr, y_bf, (long long)Lp * C2, C2, nullptr, 0, Lp);

  // ylast[b,d,l] = leaky(Wlast * y^T + blast)  (512 x 128, K=256)
  bgemm_abT<<<dim3(Lp / 16, D / 128, B), 256, 0, stream>>>(
      wlast_bf, 0, C2, y_bf, (long long)Lp * C2, C2,
      ylast, nullptr, (long long)D * Lp, Lp, blast, 1, C2);
  bn_stats<<<D, 256, 0, stream>>>(ylast, D, Lp, L, B, mean_l, inv_l);

  { size_t n = (size_t)L * B * D;
    final_out<<<gsz(n), 256, 0, stream>>>(ylast, mean_l, inv_l, l_gamma, l_beta,
                                          enc_f32, out, B, D, Lp, n); }
}
